// SpectralFloodWalkV0_6760278524243
// MI455X (gfx1250) — compile-verified, matrix operand order checked by reading, not run on hardware
//
#include <hip/hip_runtime.h>
#include <hip/hip_bf16.h>

// ---------------------------------------------------------------- constants
#define B_  16
#define S_  1024
#define C_  8
#define V_  50257
#define D_  512
#define E_  8
#define H_  512
#define R_  32
#define F_  1024
#define RT_ 512
#define CD_ 64
#define Q_  128
#define SBLK 128

typedef __attribute__((ext_vector_type(16))) _Float16 v16h;
typedef __attribute__((ext_vector_type(8)))  _Float16 v8h;
typedef __attribute__((ext_vector_type(8)))  float    v8f;
typedef __attribute__((ext_vector_type(4)))  float    v4f;

// ------------------------------------------------ WMMA fragment helpers
// A-operand layout per lane: m = row0 + (lane&15), khalf = (lane>=16)*8,
// elements 0..7 -> k = kbase+khalf+i ; 8..15 -> k = kbase+khalf+16+(i-8).
// Both runs are 8 contiguous halfs (16B) -> two ds_load_b128.
__device__ __forceinline__ v16h frag_a_lds(const _Float16* base, int row0,
                                           int ld, int kbase, int lane) {
  const _Float16* p =
      base + (size_t)(row0 + (lane & 15)) * ld + kbase + ((lane >> 4) << 3);
  union { v16h f; v8h h[2]; } u;
  u.h[0] = *(const v8h*)p;
  u.h[1] = *(const v8h*)(p + 16);
  return u.f;
}

// fragment-layout position for a [K=32 x N] tile stored as per-(ntile,lane)
// contiguous 32B chunks:  chunk = (n>>4)*32 + (n&15) + ((k>>3)&1)*16,
// element = (k&7) + ((k>>4)<<3)
__device__ __forceinline__ int fragpos(int k, int n) {
  int lane = (n & 15) + (((k >> 3) & 1) << 4);
  int i = (k & 7) + ((k >> 4) << 3);
  return (((n >> 4) << 5) + lane) * 16 + i;
}

// load one pre-packed B fragment: 32 contiguous bytes -> two ds_load_b128
__device__ __forceinline__ v16h frag_b_pack(const _Float16* base, int ntile,
                                            int lane) {
  const _Float16* p = base + (size_t)((ntile << 5) + lane) * 16;
  union { v16h f; v8h h[2]; } u;
  u.h[0] = *(const v8h*)p;
  u.h[1] = *(const v8h*)(p + 8);
  return u.f;
}

__device__ __forceinline__ v8f wmma16(v16h a, v16h b, v8f c) {
  return __builtin_amdgcn_wmma_f32_16x16x32_f16(false, a, false, b, (short)0, c,
                                                false, false);
}

// ------------------------------------------------ generic WMMA GEMM
// out[M,N] = act(in[M,K] @ w[K,N] + bias[N]); z-batched via element strides.
// act: 0=none 1=silu 2=sigmoid.  Double-buffered LDS, 1 barrier per k-step.
__global__ __launch_bounds__(256) void sfw_gemm(
    const float* __restrict__ in, const float* __restrict__ wgt,
    const float* __restrict__ bias, float* __restrict__ out, int M, int N,
    int K, int act, long in_bs, long w_bs, long b_bs, long out_bs) {
  __shared__ __align__(16) _Float16 sA[2][64 * 32];       // row-major, 64B rows
  __shared__ __align__(16) _Float16 sBf[2][4 * 32 * 16];  // fragment layout
  const int t = threadIdx.x, lane = t & 31, wv = t >> 5;
  const int mt = wv & 3, nh = wv >> 2;  // wave -> (m-tile, n-half)
  const int bm = blockIdx.x * 64, bn = blockIdx.y * 64;
  const int e = blockIdx.z;
  in  += (size_t)e * in_bs;
  wgt += (size_t)e * w_bs;
  out += (size_t)e * out_bs;
  v8f acc0 = {}, acc1 = {};
  const int ar = t >> 2, ac = (t & 3) * 8;  // A cooperative load: 64x32
  const int br = t >> 3, bc = (t & 7) * 8;  // B cooperative load: 32x64
  const bool arow_ok = (bm + ar) < M;
  const bool nfull = (bn + 64) <= N;          // whole 64-col tile in bounds
  const bool nvec = nfull && ((N & 3) == 0);  // 16B-aligned rows -> b128 loads

  auto stageA = [&](int k0, int p) {
    v8h tmp;
    if (arow_ok) {
      const float* src = in + (size_t)(bm + ar) * K + k0 + ac;
      __builtin_prefetch(src + 32, 0, 1);  // global_prefetch_b8 (speculative)
#pragma unroll
      for (int i = 0; i < 8; ++i) tmp[i] = (_Float16)src[i];
    } else {
#pragma unroll
      for (int i = 0; i < 8; ++i) tmp[i] = (_Float16)0.f;
    }
    *(v8h*)(sA[p] + ar * 32 + ac) = tmp;
  };
  auto stageB = [&](int k0, int p) {
    const float* src = wgt + (size_t)(k0 + br) * N + bn + bc;
    if (nvec) {  // two 16B vector loads, no predication
      v4f f0 = *(const v4f*)src;
      v4f f1 = *(const v4f*)(src + 4);
#pragma unroll
      for (int i = 0; i < 4; ++i) sBf[p][fragpos(br, bc + i)] = (_Float16)f0[i];
#pragma unroll
      for (int i = 0; i < 4; ++i)
        sBf[p][fragpos(br, bc + 4 + i)] = (_Float16)f1[i];
    } else if (nfull) {  // unconditional scalar loads (clause, no exec churn)
#pragma unroll
      for (int i = 0; i < 8; ++i)
        sBf[p][fragpos(br, bc + i)] = (_Float16)src[i];
    } else {  // N-edge tile: predicated
#pragma unroll
      for (int i = 0; i < 8; ++i) {
        int col = bn + bc + i;
        float v = (col < N) ? src[i] : 0.f;
        sBf[p][fragpos(br, bc + i)] = (_Float16)v;
      }
    }
  };

  int p = 0;
  stageA(0, 0);
  stageB(0, 0);
  for (int k0 = 0; k0 < K; k0 += 32) {  // all K here are multiples of 32
    __syncthreads();
    if (k0 + 32 < K) {  // prefetch next tile into the other buffer
      stageA(k0 + 32, p ^ 1);
      stageB(k0 + 32, p ^ 1);
    }
    v16h af = frag_a_lds(sA[p], mt * 16, 32, 0, lane);
    v16h b0 = frag_b_pack(sBf[p], nh * 2, lane);
    v16h b1 = frag_b_pack(sBf[p], nh * 2 + 1, lane);
    acc0 = wmma16(af, b0, acc0);
    acc1 = wmma16(af, b1, acc1);
    p ^= 1;
  }
  // epilogue: C layout: vgpr j -> row = j + (lane>=16)*8, col = lane&15
  const int rb = (lane >> 4) * 8, nc = lane & 15;
#pragma unroll
  for (int hf = 0; hf < 2; ++hf) {
    v8f acc = hf ? acc1 : acc0;
    int col = bn + nh * 32 + hf * 16 + nc;
    if (col < N) {
      float bv = bias ? bias[(size_t)e * b_bs + col] : 0.f;
#pragma unroll
      for (int j = 0; j < 8; ++j) {
        int row = bm + mt * 16 + rb + j;
        if (row < M) {
          float v = acc[j] + bv;
          if (act == 1) v = v / (1.f + __expf(-v));
          else if (act == 2) v = 1.f / (1.f + __expf(-v));
          out[(size_t)row * N + col] = v;
        }
      }
    }
  }
}

// ------------------------------------------------ WMMA recurrent scan
// One workgroup (8 waves) per expert. State[16x512] lives in registers in the
// WMMA C-layout; each step: low = (state @ v) * g ; state = a*state + u + low @ u^T
__global__ __launch_bounds__(256) void sfw_scan(
    const float* __restrict__ Ablk, const float* __restrict__ Ublk,
    const float* __restrict__ Gblk, const float* __restrict__ v_mat,
    const float* __restrict__ u_mat, float* __restrict__ states, int sblk) {
  __shared__ __align__(16) _Float16 s_state_h[16 * 512];  // 16 KB
  __shared__ __align__(16) _Float16 s_uTf[32 * 32 * 16];  // 32 KB frag layout
  __shared__ __align__(16) _Float16 s_low[16 * 32];       // 1 KB
  const int e = blockIdx.x;
  const int t = threadIdx.x, lane = t & 31, wv = t >> 5;
  const int rb = (lane >> 4) * 8, nc = lane & 15;
  // register-resident state: (b = rb+j, h = wv*64 + nt*16 + nc)
  float st[4][8];
#pragma unroll
  for (int nt = 0; nt < 4; ++nt)
#pragma unroll
    for (int j = 0; j < 8; ++j)
      st[nt][j] =
          states[(size_t)e * 8192 + (rb + j) * 512 + wv * 64 + nt * 16 + nc];
  // u^T (B-operand of the delta matmul) into LDS fragment layout:
  // element (k=r, n=h) from u_mat[e][h][r]
  for (int i = t; i < 512 * 32; i += 256) {
    int h = i >> 5, r = i & 31;
    s_uTf[fragpos(r, h)] = (_Float16)u_mat[(size_t)e * 16384 + i];
  }
  // v fragments pre-staged in registers (waves 0,1 own low's two n-tiles)
  v16h vfrag[16];
  if (wv < 2) {
    const float* vp = v_mat + (size_t)e * 16384;  // [H][R]
    const int kh = (lane >> 4) << 3;
    const int n = wv * 16 + nc;
#pragma unroll
    for (int kk = 0; kk < 16; ++kk) {
#pragma unroll
      for (int i = 0; i < 8; ++i)
        vfrag[kk][i] = (_Float16)vp[(size_t)(kk * 32 + kh + i) * 32 + n];
#pragma unroll
      for (int i = 0; i < 8; ++i)
        vfrag[kk][8 + i] = (_Float16)vp[(size_t)(kk * 32 + kh + 16 + i) * 32 + n];
    }
  }
  __syncthreads();
  for (int ss = 0; ss < sblk; ++ss) {
    // publish state as f16 for the A-operand of "low"
#pragma unroll
    for (int nt = 0; nt < 4; ++nt)
#pragma unroll
      for (int j = 0; j < 8; ++j)
        s_state_h[(rb + j) * 512 + wv * 64 + nt * 16 + nc] = (_Float16)st[nt][j];
    __syncthreads();
    if (wv < 2) {  // low[16x32] = state @ v, gated by g
      v8f acc = {};
#pragma unroll
      for (int kk = 0; kk < 16; ++kk) {
        v16h a = frag_a_lds(s_state_h, 0, 512, kk * 32, lane);
        acc = wmma16(a, vfrag[kk], acc);
      }
      int rr = wv * 16 + nc;
#pragma unroll
      for (int j = 0; j < 8; ++j) {
        int bb = rb + j;
        float g = Gblk[(((size_t)e * 16 + bb) * sblk + ss) * 32 + rr];
        s_low[bb * 32 + rr] = (_Float16)(acc[j] * g);
      }
    }
    __syncthreads();
    // state = a*state + u + low @ u^T ; each wave owns 64 h-columns
    v16h af = frag_a_lds(s_low, 0, 32, 0, lane);
#pragma unroll
    for (int nt = 0; nt < 4; ++nt) {
      int h0 = wv * 64 + nt * 16;
      v16h b = frag_b_pack(s_uTf, (h0 >> 4), lane);
      v8f acc = {};
      acc = wmma16(af, b, acc);
#pragma unroll
      for (int j = 0; j < 8; ++j) {
        int bb = rb + j;
        size_t row = ((size_t)e * 16 + bb) * sblk + ss;
        float a = Ablk[row * 512 + h0 + nc];
        float u = Ublk[row * 512 + h0 + nc];
        st[nt][j] = a * st[nt][j] + u + acc[j];
      }
    }
    __syncthreads();
  }
#pragma unroll
  for (int nt = 0; nt < 4; ++nt)
#pragma unroll
    for (int j = 0; j < 8; ++j)
      states[(size_t)e * 8192 + (rb + j) * 512 + wv * 64 + nt * 16 + nc] =
          st[nt][j];
}

// ------------------------------------------------ small helper kernels
__global__ void sfw_zero(float* p, size_t n) {
  size_t i = (size_t)blockIdx.x * blockDim.x + threadIdx.x;
  if (i < n) p[i] = 0.f;
}

__global__ void sfw_embed(const int* tok, const float* emb, float* out,
                          int n_tok, int ldout) {
  size_t i = (size_t)blockIdx.x * blockDim.x + threadIdx.x;
  if (i >= (size_t)n_tok * D_) return;
  int row = (int)(i / D_), d = (int)(i % D_);
  out[(size_t)row * ldout + d] = emb[(size_t)tok[row] * D_ + d];
}

__global__ void sfw_embed_mean(const int* tok, const float* emb, float* out,
                               int rows, int L, int ldout, int off) {
  int i = blockIdx.x * blockDim.x + threadIdx.x;
  if (i >= rows * D_) return;
  int r = i / D_, d = i % D_;
  float s = 0.f;
  for (int l = 0; l < L; ++l) s += emb[(size_t)tok[r * L + l] * D_ + d];
  out[(size_t)r * ldout + off + d] = s / (float)L;
}

__global__ void sfw_copy_sblk(const float* X, float* XB, int s0, int sblk) {
  size_t i = (size_t)blockIdx.x * blockDim.x + threadIdx.x;
  if (i >= (size_t)B_ * sblk * D_) return;
  int d = (int)(i % D_);
  int rs = (int)(i / D_);
  int b = rs / sblk, ss = rs % sblk;
  XB[i] = X[((size_t)b * S_ + s0 + ss) * D_ + d];
}

__global__ void sfw_softmax8(const float* in, float* out) {
  int b = threadIdx.x;
  if (b >= B_) return;
  float v[E_], m = -1e30f;
  for (int e = 0; e < E_; ++e) { v[e] = in[b * E_ + e]; m = fmaxf(m, v[e]); }
  float s = 0.f;
  for (int e = 0; e < E_; ++e) { v[e] = __expf(v[e] - m); s += v[e]; }
  for (int e = 0; e < E_; ++e) out[b * E_ + e] = v[e] / s;
}

__global__ void sfw_fuse(const float* proj, const float* gate, float* fused) {
  int i = blockIdx.x * blockDim.x + threadIdx.x;
  if (i >= B_ * F_) return;
  int b = i / F_, f = i % F_;
  float s = 0.f;
  for (int e = 0; e < E_; ++e)
    s += gate[b * E_ + e] * proj[((size_t)e * B_ + b) * F_ + f];
  fused[i] = s;
}

// normalize row; if quant: straight-through int8 forward = clip(round(x̂*127))/127
__global__ void sfw_rownorm(const float* in, float* out, int cols, int quant) {
  int r = blockIdx.x, t = threadIdx.x;
  __shared__ float red[256];
  float s = 0.f;
  for (int c = t; c < cols; c += 256) { float x = in[r * cols + c]; s += x * x; }
  red[t] = s;
  __syncthreads();
  for (int k = 128; k > 0; k >>= 1) {
    if (t < k) red[t] += red[t + k];
    __syncthreads();
  }
  float n = fmaxf(sqrtf(red[0]), 1e-6f);
  for (int c = t; c < cols; c += 256) {
    float xn = in[r * cols + c] / n;
    if (quant) {
      float q = fminf(fmaxf(rintf(xn * 127.f), -127.f), 127.f);
      xn = q / 127.f;
    }
    out[r * cols + c] = xn;
  }
}

__global__ void sfw_build_cin(const float* fused, float* cin) {
  int i = blockIdx.x * blockDim.x + threadIdx.x;
  if (i >= B_ * 2 * F_) return;
  int b = i >> 11, c = i & 2047;
  cin[i] = (c < F_) ? fused[b * F_ + c] : 0.f;
}

__global__ void sfw_ctxrep(const float* ctx, float* hin) {
  int i = blockIdx.x * blockDim.x + threadIdx.x;
  if (i >= B_ * C_ * F_) return;
  int row = i >> 10, f = i & 1023;
  hin[(size_t)row * (D_ + F_) + D_ + f] = ctx[(row >> 3) * F_ + f];
}

__global__ void sfw_addpos(float* h0, const float* chunk_pos) {
  int i = blockIdx.x * blockDim.x + threadIdx.x;
  if (i >= B_ * C_ * F_) return;
  int row = i >> 10, c = row & (C_ - 1);
  h0[i] += chunk_pos[(c << 10) + (i & 1023)];
}

__global__ void sfw_nll(const float* logits, const int* targets,
                        float* rowloss) {
  int r = blockIdx.x, t = threadIdx.x;
  __shared__ float red[256];
  const float* lp = logits + (size_t)r * V_;
  float m = -1e30f;
  for (int c = t; c < V_; c += 256) m = fmaxf(m, lp[c]);
  red[t] = m;
  __syncthreads();
  for (int k = 128; k > 0; k >>= 1) {
    if (t < k) red[t] = fmaxf(red[t], red[t + k]);
    __syncthreads();
  }
  m = red[0];
  __syncthreads();
  float s = 0.f;
  for (int c = t; c < V_; c += 256) s += __expf(lp[c] - m);
  red[t] = s;
  __syncthreads();
  for (int k = 128; k > 0; k >>= 1) {
    if (t < k) red[t] += red[t + k];
    __syncthreads();
  }
  if (t == 0) rowloss[r] = logf(red[0]) + m - lp[targets[r]];
}

__global__ void sfw_mean128(const float* in, float* out) {
  __shared__ float red[128];
  int t = threadIdx.x;
  red[t] = in[t];
  __syncthreads();
  for (int k = 64; k > 0; k >>= 1) {
    if (t < k) red[t] += red[t + k];
    __syncthreads();
  }
  if (t == 0) *out = red[0] / 128.f;
}

__global__ void sfw_sim(const float* q, const float* kst, float* out) {
  int i = blockIdx.x * blockDim.x + threadIdx.x;
  if (i >= B_ * B_) return;
  int r = i >> 4, c = i & 15;
  float s = 0.f;
  for (int d = 0; d < Q_; ++d) s += q[r * Q_ + d] * kst[c * Q_ + d];
  out[i] = s * 0.08838834764831845f;  // 1/sqrt(128)
}

// ---------------------------------------------------------------- launch
extern "C" void kernel_launch(void* const* d_in, const int* in_sizes, int n_in,
                              void* d_out, int out_size, void* d_ws,
                              size_t ws_size, hipStream_t stream) {
  (void)in_sizes; (void)n_in; (void)out_size; (void)ws_size;
  const int* prefix_tokens = (const int*)d_in[0];
  const int* chunk_inputs  = (const int*)d_in[1];
  const int* chunk_targets = (const int*)d_in[2];
  const int* next_tokens   = (const int*)d_in[3];
  const float* emb  = (const float*)d_in[4];
  const float* Wa = (const float*)d_in[5];  const float* ba = (const float*)d_in[6];
  const float* Wg = (const float*)d_in[7];  const float* bg = (const float*)d_in[8];
  const float* Wu = (const float*)d_in[9];  const float* bu = (const float*)d_in[10];
  const float* u_mat = (const float*)d_in[11];
  const float* v_mat = (const float*)d_in[12];
  const float* Wp = (const float*)d_in[13]; const float* bp = (const float*)d_in[14];
  const float* Wg1 = (const float*)d_in[15]; const float* bg1 = (const float*)d_in[16];
  const float* Wg2 = (const float*)d_in[17]; const float* bg2 = (const float*)d_in[18];
  const float* Wq = (const float*)d_in[19];  const float* bq = (const float*)d_in[20];
  const float* Wt1 = (const float*)d_in[21]; const float* bt1 = (const float*)d_in[22];
  const float* Wt2 = (const float*)d_in[23]; const float* bt2 = (const float*)d_in[24];
  const float* Wk = (const float*)d_in[25];  const float* bk = (const float*)d_in[26];
  const float* Web = (const float*)d_in[27];
  const float* Wwc = (const float*)d_in[28]; const float* bwc = (const float*)d_in[29];
  const float* Wc1 = (const float*)d_in[30]; const float* bc1 = (const float*)d_in[31];
  const float* Wc2 = (const float*)d_in[32]; const float* bc2 = (const float*)d_in[33];
  const float* chunk_pos = (const float*)d_in[34];
  const float* Wci = (const float*)d_in[35]; const float* bci = (const float*)d_in[36];
  const float* Wm1 = (const float*)d_in[37]; const float* bm1 = (const float*)d_in[38];
  const float* Wm2 = (const float*)d_in[39]; const float* bm2 = (const float*)d_in[40];
  const float* Wlm = (const float*)d_in[41]; const float* blm = (const float*)d_in[42];
  float* out = (float*)d_out;

  // workspace carve-up (fp32 elements)
  float* wsf = (float*)d_ws;
  size_t o = 0;
  auto carve = [&](size_t n) { float* p = wsf + o; o += n; return p; };
  float* X      = carve((size_t)B_ * S_ * D_);
  float* XBLK   = carve((size_t)B_ * SBLK * D_);
  float* ABLK   = carve((size_t)E_ * B_ * SBLK * H_);
  float* UBLK   = carve((size_t)E_ * B_ * SBLK * H_);
  float* GBLK   = carve((size_t)E_ * B_ * SBLK * R_);
  float* STATES = carve((size_t)E_ * B_ * H_);
  float* SUM    = carve(B_ * D_);
  float* GH1    = carve(B_ * D_);
  float* GLOG   = carve(B_ * E_);
  float* GATE   = carve(B_ * E_);
  float* PROJ   = carve((size_t)E_ * B_ * F_);
  float* FUSED  = carve(B_ * F_);
  float* FQ     = carve(B_ * Q_);
  float* QST    = carve(B_ * Q_);
  float* TIN    = carve(B_ * 2 * D_);
  float* T1     = carve(B_ * RT_);
  float* T2     = carve(B_ * RT_);
  float* TST    = carve(B_ * RT_);
  float* TK     = carve(B_ * Q_);
  float* TKST   = carve(B_ * Q_);
  float* CIN    = carve(B_ * 2 * F_);
  float* CH     = carve(B_ * F_);
  float* CTX    = carve(B_ * F_);
  float* WCODES = carve(B_ * CD_);
  float* WRAW   = carve(B_ * RT_);
  float* WST    = carve(B_ * RT_);
  float* WK     = carve(B_ * Q_);
  float* HIN    = carve((size_t)B_ * C_ * (D_ + F_));
  float* H0     = carve((size_t)B_ * C_ * F_);
  float* H1     = carve((size_t)B_ * C_ * F_);
  float* HID    = carve((size_t)B_ * C_ * F_);
  float* RLOSS  = carve(B_ * C_);

  auto gemm = [&](const float* in, const float* wgt, const float* bias,
                  float* dst, int M, int N, int K, int act, int batch,
                  long in_bs, long w_bs, long b_bs, long out_bs) {
    dim3 g((M + 63) / 64, (N + 63) / 64, batch);
    sfw_gemm<<<g, 256, 0, stream>>>(in, wgt, bias, dst, M, N, K, act, in_bs,
                                    w_bs, b_bs, out_bs);
  };
  auto blocks = [](size_t n) { return (unsigned)((n + 255) / 256); };

  // ---- encode prefix: gather + summary ----
  sfw_embed<<<blocks((size_t)B_ * S_ * D_), 256, 0, stream>>>(prefix_tokens,
                                                              emb, X, B_ * S_,
                                                              D_);
  sfw_embed_mean<<<blocks(B_ * D_), 256, 0, stream>>>(prefix_tokens, emb, SUM,
                                                      B_, S_, D_, 0);
  // gate MLP + softmax
  gemm(SUM, Wg1, bg1, GH1, B_, D_, D_, 1, 1, 0, 0, 0, 0);
  gemm(GH1, Wg2, bg2, GLOG, B_, E_, D_, 0, 1, 0, 0, 0, 0);
  sfw_softmax8<<<1, 32, 0, stream>>>(GLOG, GATE);

  // ---- expert scan, blocked over S ----
  sfw_zero<<<blocks((size_t)E_ * B_ * H_), 256, 0, stream>>>(
      STATES, (size_t)E_ * B_ * H_);
  for (int s0 = 0; s0 < S_; s0 += SBLK) {
    sfw_copy_sblk<<<blocks((size_t)B_ * SBLK * D_), 256, 0, stream>>>(X, XBLK,
                                                                      s0, SBLK);
    long mrows = (long)B_ * SBLK;
    gemm(XBLK, Wa, ba, ABLK, (int)mrows, H_, D_, 2, E_, 0, (long)D_ * H_, H_,
         mrows * H_);
    gemm(XBLK, Wu, bu, UBLK, (int)mrows, H_, D_, 0, E_, 0, (long)D_ * H_, H_,
         mrows * H_);
    gemm(XBLK, Wg, bg, GBLK, (int)mrows, R_, D_, 0, E_, 0, (long)D_ * R_, R_,
         mrows * R_);
    sfw_scan<<<E_, 256, 0, stream>>>(ABLK, UBLK, GBLK, v_mat, u_mat, STATES,
                                     SBLK);
  }
  // per-expert output projection, gate-weighted fuse
  gemm(STATES, Wp, bp, PROJ, B_, F_, H_, 0, E_, (long)B_ * H_, (long)H_ * F_,
       F_, (long)B_ * F_);
  sfw_fuse<<<blocks(B_ * F_), 256, 0, stream>>>(PROJ, GATE, FUSED);

  // ---- query ----
  gemm(FUSED, Wq, bq, FQ, B_, Q_, F_, 0, 1, 0, 0, 0, 0);
  sfw_rownorm<<<B_, 256, 0, stream>>>(FQ, QST, Q_, 1);

  // ---- target / key ----
  sfw_embed_mean<<<blocks(B_ * D_), 256, 0, stream>>>(chunk_targets, emb, TIN,
                                                      B_, C_, 2 * D_, 0);
  sfw_embed_mean<<<blocks(B_ * D_), 256, 0, stream>>>(next_tokens, emb, TIN, B_,
                                                      C_, 2 * D_, D_);
  gemm(TIN, Wt1, bt1, T1, B_, RT_, 2 * D_, 1, 1, 0, 0, 0, 0);
  gemm(T1, Wt2, bt2, T2, B_, RT_, RT_, 0, 1, 0, 0, 0, 0);
  sfw_rownorm<<<B_, 256, 0, stream>>>(T2, TST, RT_, 0);
  gemm(TST, Wk, bk, TK, B_, Q_, RT_, 0, 1, 0, 0, 0, 0);
  sfw_rownorm<<<B_, 256, 0, stream>>>(TK, TKST, Q_, 1);

  // ---- read context (zero-read path) ----
  sfw_build_cin<<<blocks(B_ * 2 * F_), 256, 0, stream>>>(FUSED, CIN);
  gemm(CIN, Wc1, bc1, CH, B_, F_, 2 * F_, 1, 1, 0, 0, 0, 0);
  gemm(CH, Wc2, bc2, CTX, B_, F_, F_, 0, 1, 0, 0, 0, 0);

  // ---- write_from_ctx -> write_k_st (into d_out tail) ----
  gemm(CTX, Wwc, bwc, WCODES, B_, CD_, F_, 0, 1, 0, 0, 0, 0);
  gemm(WCODES, Web, nullptr, WRAW, B_, RT_, CD_, 0, 1, 0, 0, 0, 0);
  sfw_rownorm<<<B_, 256, 0, stream>>>(WRAW, WST, RT_, 0);
  gemm(WST, Wk, bk, WK, B_, Q_, RT_, 0, 1, 0, 0, 0, 0);

  // ---- chunk decoder ----
  sfw_embed<<<blocks((size_t)B_ * C_ * D_), 256, 0, stream>>>(
      chunk_inputs, emb, HIN, B_ * C_, D_ + F_);
  sfw_ctxrep<<<blocks((size_t)B_ * C_ * F_), 256, 0, stream>>>(CTX, HIN);
  gemm(HIN, Wci, bci, H0, B_ * C_, F_, D_ + F_, 0, 1, 0, 0, 0, 0);
  sfw_addpos<<<blocks((size_t)B_ * C_ * F_), 256, 0, stream>>>(H0, chunk_pos);
  gemm(H0, Wm1, bm1, H1, B_ * C_, F_, F_, 1, 1, 0, 0, 0, 0);
  gemm(H1, Wm2, bm2, HID, B_ * C_, F_, F_, 0, 1, 0, 0, 0, 0);
  gemm(HID, Wlm, blm, out, B_ * C_, V_, F_, 0, 1, 0, 0, 0, 0);  // logits

  // ---- losses / outputs ----
  const size_t LOGN = (size_t)B_ * C_ * V_;
  sfw_nll<<<B_ * C_, 256, 0, stream>>>(out, chunk_targets, RLOSS);
  sfw_mean128<<<1, 128, 0, stream>>>(RLOSS, out + LOGN);
  sfw_sim<<<1, 256, 0, stream>>>(QST, TKST, out + LOGN + 1);
  sfw_rownorm<<<B_, 256, 0, stream>>>(WK, out + LOGN + 1 + B_ * B_, Q_, 1);
}
